// MyGRU_GAT_12008728559868
// MI455X (gfx1250) — matrix operand order repaired and checked by reading, not compiled, not run on hardware
//
#include <hip/hip_runtime.h>
#include <hip/hip_bf16.h>
#include <math.h>

// ---------------------------------------------------------------------------
// Shapes (fixed by the reference)
// ---------------------------------------------------------------------------
#define Bb 8
#define Tt 64
#define Ss (Bb * Tt)      // 512 positions
#define Nn 32             // nodes per subgraph
#define Ee 256            // edges per subgraph
#define Dd 512            // embed dim
#define Hh 8              // heads
#define DHh 64            // head dim
#define Vv 50000          // output vocab
#define CIi 1024          // 2*D

typedef __attribute__((ext_vector_type(16))) __bf16 v16bf;
typedef __attribute__((ext_vector_type(8)))  float  v8f;
typedef __attribute__((ext_vector_type(4)))  unsigned int u32x4;

union FragBF  { v16bf v; u32x4 q[2]; __bf16 h[16]; };

// pointer types matching the async-LDS builtin prototype:
//   param0: int vector_size(16), address space 1 (global / "__device__")
//   param1: int vector_size(16), address space 3 (LDS / "__shared__")
typedef int v4i_vs __attribute__((vector_size(16)));
typedef __attribute__((address_space(1))) v4i_vs v4i_gbl;
typedef __attribute__((address_space(3))) v4i_vs v4i_lds;

// async global->LDS staging path (gfx1250 ASYNCcnt ops), with sync fallback
#if defined(__has_builtin)
#  if __has_builtin(__builtin_amdgcn_global_load_async_to_lds_b128)
#    define HAVE_ASYNC_LDS 1
#  endif
#endif

__device__ __forceinline__ void wait_asynccnt0() {
#if defined(__has_builtin) && __has_builtin(__builtin_amdgcn_s_wait_asynccnt)
    __builtin_amdgcn_s_wait_asynccnt(0);
#else
    asm volatile("s_wait_asynccnt 0x0" ::: "memory");
#endif
}

// ---------------------------------------------------------------------------
// Generic bf16 WMMA GEMM:  C[M,N] = A[M,K] * B[K,N]  (+ bias)
// A, B fp32 in memory; A async-staged into LDS (fp32) then converted at
// fragment build; B converted to bf16 while staging (transposed) into LDS.
// Optional rowidx: A row m is read from A[rowidx[m]] (gathered embedding GEMM).
// Block: 256 threads = 8 waves; tile BM=64, BN=128, BK=32.
// Wave (wm in 0..3, wn in 0..1) computes a 16x64 patch as 4 WMMA accumulators.
// ---------------------------------------------------------------------------
#define BM 64
#define BN 128
#define BK 32

__global__ __launch_bounds__(256) void gemm_bf16_wmma(
    const float* __restrict__ A, const float* __restrict__ B,
    float* __restrict__ C, const float* __restrict__ bias,
    const int* __restrict__ rowidx,
    int M, int N, int K, int lda, int ldb, int ldc)
{
    __shared__ float  Asf[BM][BK];      // 8 KB fp32 (row stride 128 B)
    __shared__ __bf16 Bs[BN][BK];       // 8 KB bf16 transposed: Bs[n][k]

    const int tid  = threadIdx.x;
    const int lane = tid & 31;
    const int wid  = tid >> 5;
    const int wm   = wid >> 1;          // 0..3
    const int wn   = wid & 1;           // 0..1
    const int mblk = blockIdx.y * BM;
    const int nblk = blockIdx.x * BN;

    const v8f vzero = {0.f, 0.f, 0.f, 0.f, 0.f, 0.f, 0.f, 0.f};
    v8f acc[4] = {vzero, vzero, vzero, vzero};

    // A staging map: thread -> rows (ar, ar+32), 4 consecutive k at ac
    const int ar = tid >> 3;            // 0..31
    const int ac = (tid & 7) * 4;       // 0,4,...,28
    const int gm0 = mblk + ar;
    const int gm1 = mblk + ar + 32;
    const int as0 = rowidx ? rowidx[gm0] : gm0;
    const int as1 = rowidx ? rowidx[gm1] : gm1;
    const float* __restrict__ Ap0 = A + (size_t)as0 * lda + ac;
    const float* __restrict__ Ap1 = A + (size_t)as1 * lda + ac;

    // B staging map: thread owns column n = tid&127, k-half (tid>>7)*16
    const int bn  = tid & (BN - 1);
    const int bkh = (tid >> 7) * 16;
    const int gn_st = nblk + bn;
    const float* __restrict__ Bcol = B + gn_st;

    const int laneM = lane & 15;
    const int kbA   = (lane >> 4) * 8;   // A frag: element offset 0 or 8
    const int kbB   = (lane >> 4) * 8;   // B frag: element offset 0 or 8

    for (int k0 = 0; k0 < K; k0 += BK) {
        // ---- stage A tile: two b128 global->LDS transfers per thread ----
#if defined(HAVE_ASYNC_LDS)
        __builtin_amdgcn_global_load_async_to_lds_b128(
            (v4i_gbl*)(Ap0 + k0), (v4i_lds*)&Asf[ar][ac], 0, 0);
        __builtin_amdgcn_global_load_async_to_lds_b128(
            (v4i_gbl*)(Ap1 + k0), (v4i_lds*)&Asf[ar + 32][ac], 0, 0);
#else
        {
            const float4 a0 = *(const float4*)(Ap0 + k0);
            const float4 a1 = *(const float4*)(Ap1 + k0);
            *(float4*)&Asf[ar][ac]      = a0;
            *(float4*)&Asf[ar + 32][ac] = a1;
        }
#endif

        // ---- stage B tile transposed (convert fp32 -> bf16) ----
        if (gn_st < N) {
            const float* bp = Bcol + (size_t)(k0 + bkh) * ldb;
            __builtin_prefetch(bp + (size_t)BK * ldb, 0, 1);  // global_prefetch_b8
            #pragma unroll
            for (int kk = 0; kk < 16; ++kk)
                Bs[bn][bkh + kk] = (__bf16)bp[(size_t)kk * ldb];
        } else {
            #pragma unroll
            for (int kk = 0; kk < 16; ++kk)
                Bs[bn][bkh + kk] = (__bf16)0.0f;
        }

#if defined(HAVE_ASYNC_LDS)
        wait_asynccnt0();
#endif
        __syncthreads();

        // ---- A fragment: fp32 LDS -> packed bf16 (cvt_pk at build time) ----
        // lanes 0-15 hold K 0..7 / 16..23, lanes 16-31 hold 8..15 / 24..31
        FragBF fa;
        const float* arow = &Asf[wm * 16 + laneM][0];
        #pragma unroll
        for (int i = 0; i < 8; ++i) fa.h[i]     = (__bf16)arow[kbA + i];
        #pragma unroll
        for (int i = 0; i < 8; ++i) fa.h[8 + i] = (__bf16)arow[kbA + 16 + i];

        #pragma unroll
        for (int t = 0; t < 4; ++t) {
            // B fragment: lane holds B[kbB+0..7, n] and B[kbB+16..23, n] from Bs[n][...]
            FragBF fb;
            const int nloc = wn * 64 + t * 16 + laneM;
            fb.q[0] = *(const u32x4*)&Bs[nloc][kbB];
            fb.q[1] = *(const u32x4*)&Bs[nloc][kbB + 16];
            acc[t] = __builtin_amdgcn_wmma_f32_16x16x32_bf16(
                false, fa.v, false, fb.v, (short)0, acc[t], false, false);
        }
        __syncthreads();
    }

    // ---- store C (f32 C/D layout: v_i -> M = i + 8*(lane>=16), N = lane&15) ----
    #pragma unroll
    for (int t = 0; t < 4; ++t) {
        const int n = nblk + wn * 64 + t * 16 + laneM;
        if (n >= N) continue;
        const float badd = bias ? bias[n] : 0.0f;
        #pragma unroll
        for (int i = 0; i < 8; ++i) {
            const int m = mblk + wm * 16 + (lane >> 4) * 8 + i;
            C[(size_t)m * ldc + n] = acc[t][i] + badd;
        }
    }
}

// ---------------------------------------------------------------------------
// GAT attention for target node 0 only (the only node the model consumes).
// One block per position s. Produces inp[s] = [ X[idx[s,0]] | gat_out0 + b ].
// ---------------------------------------------------------------------------
__global__ __launch_bounds__(256) void gat_attn_kernel(
    const int* __restrict__ xidx,      // [S*N]
    const int* __restrict__ eidx,      // [S*2*E]
    const float* __restrict__ X,       // [NV, D]
    const float* __restrict__ xW,      // [S*N, D]
    const float* __restrict__ att_src, // [H, DH]
    const float* __restrict__ att_dst, // [H, DH]
    const float* __restrict__ b_gat,   // [D]
    float* __restrict__ inp)           // [S, 2D]
{
    const int s   = blockIdx.x;
    const int tid = threadIdx.x;

    __shared__ float a_s[Nn][Hh];
    __shared__ float a_d0[Hh];
    __shared__ float esc[Ee + 1][Hh];  // scores -> alpha (in place)
    __shared__ int   ssrc[Ee + 1];
    __shared__ float mh[Hh], den[Hh];

    const float* __restrict__ xWs = xW + (size_t)s * Nn * Dd;

    // per-node/head attention logits a_s; a_d only needed for node 0
    {
        const int n = tid >> 3, h = tid & 7;
        const float* row = xWs + n * Dd + h * DHh;
        const float* asp = att_src + h * DHh;
        float acc = 0.0f;
        #pragma unroll 8
        for (int d = 0; d < DHh; ++d) acc += row[d] * asp[d];
        a_s[n][h] = acc;
        if (n == 0) {
            const float* adp = att_dst + h * DHh;
            float accd = 0.0f;
            #pragma unroll 8
            for (int d = 0; d < DHh; ++d) accd += row[d] * adp[d];
            a_d0[h] = accd;
        }
    }
    __syncthreads();

    // leaky_relu scores for edges targeting node 0 (+ self loop at slot Ee)
    {
        const int e    = tid;
        const int srcn = eidx[(size_t)s * 2 * Ee + e];
        const int dstn = eidx[(size_t)s * 2 * Ee + Ee + e];
        const bool act = (dstn == 0);
        ssrc[e] = act ? srcn : -1;
        #pragma unroll
        for (int h = 0; h < Hh; ++h) {
            float v = -1e30f;
            if (act) {
                const float x = a_s[srcn][h] + a_d0[h];
                v = (x > 0.0f) ? x : 0.2f * x;
            }
            esc[e][h] = v;
        }
        if (tid == 0) {
            ssrc[Ee] = 0;  // self loop (0 -> 0)
            #pragma unroll
            for (int h = 0; h < Hh; ++h) {
                const float x = a_s[0][h] + a_d0[h];
                esc[Ee][h] = (x > 0.0f) ? x : 0.2f * x;
            }
        }
    }
    __syncthreads();

    // deterministic per-head max + sum
    if (tid < Hh) {
        const int h = tid;
        float m = -1e30f;
        for (int e = 0; e <= Ee; ++e)
            if (ssrc[e] >= 0 && esc[e][h] > m) m = esc[e][h];
        float d = 0.0f;
        for (int e = 0; e <= Ee; ++e)
            if (ssrc[e] >= 0) d += expf(esc[e][h] - m);
        mh[h] = m;
        den[h] = d + 1e-16f;
    }
    __syncthreads();

    // scores -> alpha
    for (int e = tid; e <= Ee; e += 256) {
        #pragma unroll
        for (int h = 0; h < Hh; ++h)
            esc[e][h] = (ssrc[e] >= 0) ? expf(esc[e][h] - mh[h]) / den[h] : 0.0f;
    }
    __syncthreads();

    // aggregate and emit concatenated GRU input
    const int x0row = xidx[(size_t)s * Nn];
    for (int d = tid; d < Dd; d += 256) {
        const int h = d >> 6;
        float acc = 0.0f;
        for (int e = 0; e <= Ee; ++e) {
            const int sr = ssrc[e];
            if (sr >= 0) acc += esc[e][h] * xWs[sr * Dd + d];
        }
        inp[(size_t)s * CIi + d]      = X[(size_t)x0row * Dd + d];
        inp[(size_t)s * CIi + Dd + d] = acc + b_gat[d];
    }
}

// ---------------------------------------------------------------------------
// Sequential 2-layer GRU scan. One workgroup, 1024 threads, state in LDS.
// uproj[s] = [ u@Wz1 | u@Wr1 | u@W1 ] precomputed by WMMA GEMMs.
// ---------------------------------------------------------------------------
__global__ __launch_bounds__(1024) void gru_seq_kernel(
    const float* __restrict__ uproj,   // [S, 3*D]
    const float* __restrict__ Uz1, const float* __restrict__ Ur1,
    const float* __restrict__ U1,  const float* __restrict__ bW1,
    const float* __restrict__ bU1,
    const float* __restrict__ Wz2, const float* __restrict__ Uz2,
    const float* __restrict__ Wr2, const float* __restrict__ Ur2,
    const float* __restrict__ W2,  const float* __restrict__ bW2,
    const float* __restrict__ U2,  const float* __restrict__ bU2,
    float* __restrict__ H2out)         // [S, D]
{
    __shared__ float h1[Dd], h2[Dd], z1[Dd], r1[Dd], h1n[Dd], z2[Dd], r2[Dd];
    const int tid = threadIdx.x;
    const int j   = tid & (Dd - 1);
    const bool lo = (tid < Dd);

    if (lo) { h1[j] = 0.0f; h2[j] = 0.0f; }
    __syncthreads();

    for (int s = 0; s < Ss; ++s) {
        const float* up = uproj + (size_t)s * (3 * Dd);

        // Phase A: z1 (threads 0..511), r1 (threads 512..1023)
        {
            const float* U = lo ? Uz1 : Ur1;
            float acc = lo ? up[j] : up[Dd + j];
            for (int k = 0; k < Dd; ++k) acc += h1[k] * U[(size_t)k * Dd + j];
            const float sv = 1.0f / (1.0f + expf(-acc));
            if (lo) z1[j] = sv; else r1[j] = sv;
        }
        __syncthreads();

        // Phase B: candidate + h1 update
        if (lo) {
            float acc = up[2 * Dd + j] + bW1[j] + bU1[j];
            for (int k = 0; k < Dd; ++k) acc += (r1[k] * h1[k]) * U1[(size_t)k * Dd + j];
            const float ht1 = tanhf(acc);
            h1n[j] = z1[j] * ht1 + (1.0f - z1[j]) * h1[j];
        }
        __syncthreads();

        // Phase C: z2 / r2 from h1n and previous h2
        {
            const float* W = lo ? Wz2 : Wr2;
            const float* U = lo ? Uz2 : Ur2;
            float acc = 0.0f;
            for (int k = 0; k < Dd; ++k)
                acc += h1n[k] * W[(size_t)k * Dd + j] + h2[k] * U[(size_t)k * Dd + j];
            const float sv = 1.0f / (1.0f + expf(-acc));
            if (lo) z2[j] = sv; else r2[j] = sv;
        }
        __syncthreads();

        // Phase D: layer-2 candidate, h2 update, emit
        float h2n = 0.0f, h1v = 0.0f;
        if (lo) {
            float acc = bW2[j] + bU2[j];
            for (int k = 0; k < Dd; ++k)
                acc += h1n[k] * W2[(size_t)k * Dd + j] + (r2[k] * h2[k]) * U2[(size_t)k * Dd + j];
            const float ht2 = tanhf(acc);
            h2n = z2[j] * ht2 + (1.0f - z2[j]) * h2[j];
            h1v = h1n[j];
            H2out[(size_t)s * Dd + j] = h2n;
        }
        __syncthreads();
        if (lo) { h2[j] = h2n; h1[j] = h1v; }
        __syncthreads();
    }
}

// ---------------------------------------------------------------------------
// Row-wise log_softmax over V columns, in place on d_out.
// ---------------------------------------------------------------------------
__global__ __launch_bounds__(256) void logsoftmax_kernel(float* __restrict__ out, int V)
{
    const int r   = blockIdx.x;
    const int tid = threadIdx.x;
    float* row = out + (size_t)r * V;
    __shared__ float red[256];

    float m = -1e30f;
    for (int c = tid; c < V; c += 256) m = fmaxf(m, row[c]);
    red[tid] = m;
    __syncthreads();
    for (int o = 128; o > 0; o >>= 1) {
        if (tid < o) red[tid] = fmaxf(red[tid], red[tid + o]);
        __syncthreads();
    }
    const float rowmax = red[0];
    __syncthreads();

    float ssum = 0.0f;
    for (int c = tid; c < V; c += 256) ssum += expf(row[c] - rowmax);
    red[tid] = ssum;
    __syncthreads();
    for (int o = 128; o > 0; o >>= 1) {
        if (tid < o) red[tid] += red[tid + o];
        __syncthreads();
    }
    const float lz = logf(red[0]);
    __syncthreads();

    for (int c = tid; c < V; c += 256) row[c] = row[c] - rowmax - lz;
}

// ---------------------------------------------------------------------------
extern "C" void kernel_launch(void* const* d_in, const int* in_sizes, int n_in,
                              void* d_out, int out_size, void* d_ws, size_t ws_size,
                              hipStream_t stream)
{
    const int*   xidx    = (const int*)  d_in[0];
    const int*   eidx    = (const int*)  d_in[1];
    const float* X       = (const float*)d_in[2];
    const float* W_gat   = (const float*)d_in[3];
    const float* att_src = (const float*)d_in[4];
    const float* att_dst = (const float*)d_in[5];
    const float* b_gat   = (const float*)d_in[6];
    const float* Wz1 = (const float*)d_in[7];
    const float* Uz1 = (const float*)d_in[8];
    const float* Wr1 = (const float*)d_in[9];
    const float* Ur1 = (const float*)d_in[10];
    const float* W1  = (const float*)d_in[11];
    const float* bW1 = (const float*)d_in[12];
    const float* U1  = (const float*)d_in[13];
    const float* bU1 = (const float*)d_in[14];
    const float* Wz2 = (const float*)d_in[15];
    const float* Uz2 = (const float*)d_in[16];
    const float* Wr2 = (const float*)d_in[17];
    const float* Ur2 = (const float*)d_in[18];
    const float* W2  = (const float*)d_in[19];
    const float* bW2 = (const float*)d_in[20];
    const float* U2  = (const float*)d_in[21];
    const float* bU2 = (const float*)d_in[22];
    const float* Wg  = (const float*)d_in[23];
    const float* bg  = (const float*)d_in[24];
    float* out = (float*)d_out;

    // workspace layout (~40 MB)
    float* xW    = (float*)d_ws;                       // [S*N, D]
    float* inp   = xW    + (size_t)Ss * Nn * Dd;       // [S, 2D]
    float* uproj = inp   + (size_t)Ss * CIi;           // [S, 3D]
    float* H2    = uproj + (size_t)Ss * 3 * Dd;        // [S, D]

    // 1) gathered GEMM: xW = X[xidx] @ W_gat     [16384x512]*[512x512]
    gemm_bf16_wmma<<<dim3((Dd + BN - 1) / BN, (Ss * Nn) / BM), 256, 0, stream>>>(
        X, W_gat, xW, nullptr, xidx, Ss * Nn, Dd, Dd, Dd, Dd, Dd);

    // 2) attention for node 0 -> inp = [embed | gat]
    gat_attn_kernel<<<Ss, 256, 0, stream>>>(xidx, eidx, X, xW,
                                            att_src, att_dst, b_gat, inp);

    // 3) input-side GRU projections: uproj = inp @ [Wz1 | Wr1 | W1]
    gemm_bf16_wmma<<<dim3((Dd + BN - 1) / BN, Ss / BM), 256, 0, stream>>>(
        inp, Wz1, uproj,          nullptr, nullptr, Ss, Dd, CIi, CIi, Dd, 3 * Dd);
    gemm_bf16_wmma<<<dim3((Dd + BN - 1) / BN, Ss / BM), 256, 0, stream>>>(
        inp, Wr1, uproj + Dd,     nullptr, nullptr, Ss, Dd, CIi, CIi, Dd, 3 * Dd);
    gemm_bf16_wmma<<<dim3((Dd + BN - 1) / BN, Ss / BM), 256, 0, stream>>>(
        inp, W1,  uproj + 2 * Dd, nullptr, nullptr, Ss, Dd, CIi, CIi, Dd, 3 * Dd);

    // 4) sequential GRU scan
    gru_seq_kernel<<<1, 1024, 0, stream>>>(uproj, Uz1, Ur1, U1, bW1, bU1,
                                           Wz2, Uz2, Wr2, Ur2, W2, bW2, U2, bU2, H2);

    // 5) output projection: out = H2 @ Wg + bg   [512x512]*[512x50000]
    gemm_bf16_wmma<<<dim3((Vv + BN - 1) / BN, Ss / BM), 256, 0, stream>>>(
        H2, Wg, out, bg, nullptr, Ss, Vv, Dd, Dd, Vv, Vv);

    // 6) row-wise log_softmax in place
    logsoftmax_kernel<<<Ss, 256, 0, stream>>>(out, Vv);
}
